// Conv2dBN_fake_int8_55181739819548
// MI455X (gfx1250) — 8-bit matrix operand order repaired and checked by reading, compiled, not run on hardware
//
#include <hip/hip_runtime.h>
#include <hip/hip_bf16.h>


typedef int v8i __attribute__((ext_vector_type(8)));
typedef int v4i __attribute__((ext_vector_type(4)));
typedef int v2i __attribute__((ext_vector_type(2)));

#define CFG_B  8
#define CFG_C  64
#define CFG_H  32
#define CFG_W  32
#define CFG_O  64
#define CFG_K  576
#define CFG_L  1024
#define CFG_M  8192
#define KSTEPS 9

__global__ __launch_bounds__(256) void quant_im2col_kernel(
    const float* __restrict__ x, const float* __restrict__ sfeat,
    signed char* __restrict__ Pf)
{
    int idx = blockIdx.x * 256 + threadIdx.x;
    int row = idx / CFG_K;
    int k   = idx - row * CFG_K;
    int c   = k / 9;
    int rem = k - c * 9;
    int kh  = rem / 3;
    int kw  = rem - kh * 3;
    int b   = row >> 10;
    int sp  = row & 1023;
    int oh  = sp >> 5, ow = sp & 31;
    int ih  = oh + kh - 1;
    int iw  = ow + kw - 1;
    float v = 0.0f;
    if ((unsigned)ih < 32u && (unsigned)iw < 32u)
        v = x[(((b << 6) + c) << 10) + (ih << 5) + iw];
    float q = __builtin_rintf(v / sfeat[0]);
    q = fminf(fmaxf(q, -128.0f), 127.0f);
    Pf[idx] = (signed char)(int)q;
}

__global__ __launch_bounds__(256) void quant_pack_w_kernel(
    const float* __restrict__ w, const float* __restrict__ sw,
    int* __restrict__ Bpack)
{
    int tid  = blockIdx.x * 256 + threadIdx.x;
    int v    = tid & 7;
    int lane = (tid >> 3) & 31;
    int j    = (tid >> 8) & 3;
    int t    = tid >> 10;
    int n    = j * 16 + (lane & 15);
    int kb   = t * 64 + ((v >> 1) << 4) + ((lane >> 4) << 3) + ((v & 1) << 2);
    float s  = sw[n];
    unsigned pk = 0u;
    #pragma unroll
    for (int i = 0; i < 4; ++i) {
        float q = __builtin_rintf(w[n * CFG_K + kb + i] / s);
        q = fminf(fmaxf(q, -128.0f), 127.0f);
        pk |= ((unsigned)((int)q & 0xff)) << (8 * i);
    }
    Bpack[tid] = (int)pk;
}

__global__ __launch_bounds__(256) void lutconv_wmma_gemm_kernel(
    const signed char* __restrict__ Pf, const int* __restrict__ Bpack,
    const float* __restrict__ sfeat, const float* __restrict__ sw,
    const float* __restrict__ sact,  const float* __restrict__ bias,
    float* __restrict__ out)
{
    const int lane   = threadIdx.x & 31;
    const int wave   = threadIdx.x >> 5;
    const int wr     = wave >> 2;
    const int wc     = wave & 3;
    const int tile_m = blockIdx.x * 32 + wr * 16;
    const int tile_n = wc * 16;

    const int mrow = tile_m + (lane & 15);
    const signed char* arow = Pf + (size_t)mrow * CFG_K + ((lane >> 4) << 3);
    const v4i* bbase = (const v4i*)(Bpack + ((wc * 32 + lane) << 3));

    v8i acc = {};
    #pragma unroll
    for (int t = 0; t < KSTEPS; ++t) {
        const signed char* ap = arow + t * 64;
        v2i a0 = *(const v2i*)(ap +  0);
        v2i a1 = *(const v2i*)(ap + 16);
        v2i a2 = *(const v2i*)(ap + 32);
        v2i a3 = *(const v2i*)(ap + 48);
        v8i a;
        a[0] = a0.x; a[1] = a0.y; a[2] = a1.x; a[3] = a1.y;
        a[4] = a2.x; a[5] = a2.y; a[6] = a3.x; a[7] = a3.y;

        v4i b0 = bbase[t * 256 + 0];
        v4i b1 = bbase[t * 256 + 1];
        v8i b;
        b[0] = b0.x; b[1] = b0.y; b[2] = b0.z; b[3] = b0.w;
        b[4] = b1.x; b[5] = b1.y; b[6] = b1.z; b[7] = b1.w;

        acc = __builtin_amdgcn_wmma_i32_16x16x64_iu8(
            true, a, true, b, acc, false, false);
    }

    const float sf = sfeat[0];
    const float sa = sact[0];
    const int   n  = tile_n + (lane & 15);
    const float swn = sw[n];
    const float bn  = bias[n];
    const int mhalf = (lane >> 4) << 3;

    #pragma unroll
    for (int r = 0; r < 8; ++r) {
        int m = tile_m + r + mhalf;
        float f = (float)acc[r] * sf * swn + bn;
        f = __builtin_rintf(f / sa);
        f = fminf(fmaxf(f, -128.0f), 127.0f);
        f *= sa;
        out[((m >> 10) << 16) + (n << 10) + (m & 1023)] = f;
    }
}

extern "C" void kernel_launch(void* const* d_in, const int* in_sizes, int n_in,
                              void* d_out, int out_size, void* d_ws, size_t ws_size,
                              hipStream_t stream) {
    const float* x      = (const float*)d_in[0];
    const float* weight = (const float*)d_in[1];
    const float* sfeat  = (const float*)d_in[3];
    const float* sw     = (const float*)d_in[4];
    const float* sact   = (const float*)d_in[5];
    const float* bias   = (const float*)d_in[6];
    float* out = (float*)d_out;

    signed char* Pf = (signed char*)d_ws;
    int* Bpack = (int*)((char*)d_ws + (size_t)CFG_M * CFG_K);

    quant_im2col_kernel<<<(CFG_M * CFG_K) / 256, 256, 0, stream>>>(x, sfeat, Pf);
    quant_pack_w_kernel<<<9216 / 256, 256, 0, stream>>>(weight, sw, Bpack);
    lutconv_wmma_gemm_kernel<<<CFG_M / 32, 256, 0, stream>>>(
        Pf, Bpack, sfeat, sw, sact, bias, out);
}
